// DisentangledSelfAttention_26259430048374
// MI455X (gfx1250) — matrix operand order repaired and checked by reading, not compiled, not run on hardware
//
#include <hip/hip_runtime.h>

typedef unsigned short u16;
typedef unsigned int   u32;
typedef unsigned long long u64;

typedef __attribute__((ext_vector_type(16))) __bf16 bf16x16;
typedef __attribute__((ext_vector_type(8)))  float  f32x8;
typedef __attribute__((ext_vector_type(4)))  u32    u32x4;

// ---- TDM toolchain detection -------------------------------------------------
#if __has_include(<hip/amd_detail/amd_gfx1250_TDM.h>)
  #define HAVE_TDM  1
  #define TDM_ARGS6 1
#elif __has_builtin(__builtin_amdgcn_tensor_load_to_lds)
  #define HAVE_TDM  1
  #define TDM_ARGS6 0
#else
  #define HAVE_TDM  0
#endif

struct Frag { u32x4 lo, hi; };   // 16 bf16 = one WMMA A/B operand

static __device__ inline bf16x16 frag_cast(const Frag& f) {
    return __builtin_bit_cast(bf16x16, f);
}
static __device__ inline u16 f2bf(float f) {             // RNE f32 -> bf16
    u32 u = __builtin_bit_cast(u32, f);
    u32 r = u + 0x7FFFu + ((u >> 16) & 1u);
    return (u16)(r >> 16);
}
static __device__ inline float bf2f(u16 h) {
    u32 u = ((u32)h) << 16;
    return __builtin_bit_cast(float, u);
}
static __device__ inline f32x8 zero8() {
    return (f32x8){0.f,0.f,0.f,0.f,0.f,0.f,0.f,0.f};
}
static __device__ inline f32x8 wmma_bf16(const Frag& a, const Frag& b, f32x8 c) {
    return __builtin_amdgcn_wmma_f32_16x16x32_bf16(
        false, frag_cast(a), false, frag_cast(b), (short)0, c, false, false);
}
static __device__ inline float hmax16(float v) {  // reduce across 16-lane half
    v = fmaxf(v, __shfl_xor(v, 1));
    v = fmaxf(v, __shfl_xor(v, 2));
    v = fmaxf(v, __shfl_xor(v, 4));
    v = fmaxf(v, __shfl_xor(v, 8));
    return v;
}
static __device__ inline float hsum16(float v) {
    v += __shfl_xor(v, 1);
    v += __shfl_xor(v, 2);
    v += __shfl_xor(v, 4);
    v += __shfl_xor(v, 8);
    return v;
}

#if HAVE_TDM
typedef __attribute__((ext_vector_type(4))) int ix4;
typedef __attribute__((ext_vector_type(8))) int ix8;

// Issue one TDM 2D tile load: rows x cols (bf16 elements), row stride = stride_els.
// Tensor dims == tile dims (tile is fully in-bounds). D# layout per CDNA5 ISA ch.8.
static __device__ inline void tdm_load_tile_2d(const u16* g, void* lds,
                                               int rows, int cols, int stride_els) {
    const u64 ga = (u64)(size_t)g;
    const u32 lo = (u32)(size_t)lds;   // generic LDS addr truncates to LDS byte offset
    u32x4 g0;
    g0[0] = 1u;                                           // count=1 (valid user D#)
    g0[1] = lo;                                           // lds_addr
    g0[2] = (u32)ga;                                      // global_addr[31:0]
    g0[3] = (u32)((ga >> 32) & 0x01FFFFFFu) | (2u << 30); // global_addr[56:32] | type=2
    const u64 st = (u64)(u32)stride_els;
    ix8 g1;
    g1[0] = (int)(1u << 16);                              // data_size=2B (bf16)
    g1[1] = (int)(((u32)cols & 0xFFFFu) << 16);           // tensor_dim0[15:0]
    g1[2] = (int)((((u32)cols >> 16) & 0xFFFFu) |
                  (((u32)rows & 0xFFFFu) << 16));         // dim0[31:16] | dim1[15:0]
    g1[3] = (int)(((((u32)rows) >> 16) & 0xFFFFu) |
                  (((u32)cols & 0xFFFFu) << 16));         // dim1[31:16] | tile_dim0
    g1[4] = (int)((u32)rows & 0xFFFFu);                   // tile_dim1 (tile_dim2=0)
    g1[5] = (int)(u32)st;                                 // dim0_stride[31:0]
    g1[6] = (int)(u32)((st >> 32) & 0xFFFFu);             // dim0_stride[47:32]
    g1[7] = 0;
    const ix4 z4 = {0, 0, 0, 0};
#if TDM_ARGS6
    const ix8 z8 = {0, 0, 0, 0, 0, 0, 0, 0};
    __builtin_amdgcn_tensor_load_to_lds(g0, g1, z4, z4, z8, 0);
#else
    __builtin_amdgcn_tensor_load_to_lds(g0, g1, z4, z4, 0);
#endif
}
#endif

// ---------------- elementwise converts ----------------
__global__ void cvt_bf16_kernel(u16* __restrict__ dst, const float* __restrict__ src, int n) {
    int i = blockIdx.x * 256 + threadIdx.x;
    if (i < n) dst[i] = f2bf(src[i]);
}
// dst[N][K] = bf16(src[K][N])
__global__ void tcvt_bf16_kernel(u16* __restrict__ dst, const float* __restrict__ src, int N, int K) {
    int i = blockIdx.x * 256 + threadIdx.x;
    if (i < N * K) {
        int n = i / K, k = i % K;
        dst[i] = f2bf(src[(size_t)k * N + n]);
    }
}

// ---------------- GEMM: C[M,N] = act(A[M,K] @ Bt[N,K]^T + bias) ----------------
// A, Bt bf16 row-major. Block = 256 threads (8 waves), 128x128 tile, K-step 64.
// TDM path: double-buffered LDS tiles DMA'd by wave 0, synced by TENSORcnt+barrier.
template<bool OUT_BF16, bool RELU>
__global__ __launch_bounds__(256)
void gemm_bf16_kernel(const u16* __restrict__ A, const u16* __restrict__ Bt,
                      const float* __restrict__ bias, void* __restrict__ Cout,
                      int M, int N, int K) {
#if HAVE_TDM
    __shared__ __align__(16) u16 sA[2][128 * 64];
    __shared__ __align__(16) u16 sB[2][128 * 64];
#else
    __shared__ __align__(16) u16 sA[1][128 * 64];
    __shared__ __align__(16) u16 sB[1][128 * 64];
#endif
    const int tid  = threadIdx.x;
    const int wave = tid >> 5, lane = tid & 31;
    const int lm = lane & 15, half = lane >> 4;
    const int wm = wave & 3, wn = wave >> 2;        // 4x32 rows, 2x64 cols
    const int mblk = blockIdx.x * 128;
    const int nblk = blockIdx.y * 128;
    const int klo8 = half * 8;

    f32x8 acc[2][4];
    #pragma unroll
    for (int s = 0; s < 2; ++s)
        #pragma unroll
        for (int n = 0; n < 4; ++n) acc[s][n] = zero8();

#if HAVE_TDM
    if (wave == 0) {
        tdm_load_tile_2d(A  + (size_t)mblk * K, &sA[0][0], 128, 64, K);
        tdm_load_tile_2d(Bt + (size_t)nblk * K, &sB[0][0], 128, 64, K);
        __builtin_amdgcn_s_wait_tensorcnt(0);
    }
    __syncthreads();
    int cur = 0;
#endif

    for (int kb = 0; kb < K; kb += 64) {
#if HAVE_TDM
        const int nxt = cur ^ 1;
        if (wave == 0 && kb + 64 < K) {
            tdm_load_tile_2d(A  + (size_t)mblk * K + (kb + 64), &sA[nxt][0], 128, 64, K);
            tdm_load_tile_2d(Bt + (size_t)nblk * K + (kb + 64), &sB[nxt][0], 128, 64, K);
        }
        const u16* tA = sA[cur];
        const u16* tB = sB[cur];
#else
        {   // cooperative synchronous staging
            const int row = tid >> 1, cbase = (tid & 1) * 32;
            const u16* ga = A  + (size_t)(mblk + row) * K + kb + cbase;
            const u16* gb = Bt + (size_t)(nblk + row) * K + kb + cbase;
            #pragma unroll
            for (int c = 0; c < 4; ++c) {
                *(u32x4*)&sA[0][row*64 + cbase + c*8] = *(const u32x4*)(ga + c*8);
                *(u32x4*)&sB[0][row*64 + cbase + c*8] = *(const u32x4*)(gb + c*8);
            }
        }
        __syncthreads();
        const u16* tA = sA[0];
        const u16* tB = sB[0];
#endif
        #pragma unroll
        for (int j2 = 0; j2 < 2; ++j2) {
            Frag a[2], b[4];
            #pragma unroll
            for (int s = 0; s < 2; ++s) {
                const u16* p = &tA[(wm*32 + s*16 + lm) * 64 + j2*32];
                a[s].lo = *(const u32x4*)(p + klo8);
                a[s].hi = *(const u32x4*)(p + klo8 + 16);
            }
            #pragma unroll
            for (int n = 0; n < 4; ++n) {
                const u16* p = &tB[(wn*64 + n*16 + lm) * 64 + j2*32 + half*8];
                b[n].lo = *(const u32x4*)(p);
                b[n].hi = *(const u32x4*)(p + 16);
            }
            #pragma unroll
            for (int s = 0; s < 2; ++s)
                #pragma unroll
                for (int n = 0; n < 4; ++n)
                    acc[s][n] = wmma_bf16(a[s], b[n], acc[s][n]);
        }
        __syncthreads();
#if HAVE_TDM
        if (wave == 0 && kb + 64 < K) __builtin_amdgcn_s_wait_tensorcnt(0);
        __syncthreads();
        cur = nxt;
#endif
    }

    #pragma unroll
    for (int s = 0; s < 2; ++s) {
        #pragma unroll
        for (int n = 0; n < 4; ++n) {
            const int col = nblk + wn*64 + n*16 + lm;
            const float bv = bias ? bias[col] : 0.f;
            #pragma unroll
            for (int r = 0; r < 8; ++r) {
                const int row = mblk + wm*32 + s*16 + r + half*8;
                float v = acc[s][n][r] + bv;
                if (RELU) v = fmaxf(v, 0.f);
                if (OUT_BF16) ((u16*)Cout)[(size_t)row * N + col] = f2bf(v);
                else          ((float*)Cout)[(size_t)row * N + col] = v;
            }
        }
    }
}

// ---------------- per-head mean-centering + V transpose ----------------
// qf/kf/vf: f32 flat [64][1024][64]; muq/muk: bf16 [64][1024][64]; vt: bf16 [64][64][1024]
// NOTE: 1/sqrt(HD)=0.125 is folded into muq here (scores are bilinear).
__global__ __launch_bounds__(256)
void center_kernel(const float* __restrict__ qf, const float* __restrict__ kf,
                   const float* __restrict__ vf, u16* __restrict__ muq,
                   u16* __restrict__ muk, u16* __restrict__ vtb) {
    const int bh = blockIdx.x, tid = threadIdx.x;
    const int d = tid & 63, part = tid >> 6;
    __shared__ float pq[4][64], pk[4][64], mqs[64], mks[64];
    const float* q = qf + (size_t)bh * 65536;
    const float* k = kf + (size_t)bh * 65536;
    const float* v = vf + (size_t)bh * 65536;
    float sq = 0.f, sk = 0.f;
    for (int i = 0; i < 256; ++i) {
        const int l = part * 256 + i;
        sq += q[l*64 + d];
        sk += k[l*64 + d];
    }
    pq[part][d] = sq; pk[part][d] = sk;
    __syncthreads();
    if (tid < 64) {
        mqs[tid] = (pq[0][tid]+pq[1][tid]+pq[2][tid]+pq[3][tid]) * (1.f/1024.f);
        mks[tid] = (pk[0][tid]+pk[1][tid]+pk[2][tid]+pk[3][tid]) * (1.f/1024.f);
    }
    __syncthreads();
    for (int idx = tid; idx < 65536; idx += 256) {
        const int dd = idx & 63, ll = idx >> 6;
        muq[(size_t)bh*65536 + idx] = f2bf((q[idx] - mqs[dd]) * 0.125f);
        muk[(size_t)bh*65536 + idx] = f2bf(k[idx] - mks[dd]);
        vtb[(size_t)bh*65536 + (size_t)dd*1024 + ll] = f2bf(v[idx]);
    }
}

// ---------------- unary logits + softmax over L, scattered to uw[64][1024] ----------------
__global__ __launch_bounds__(256)
void unary_kernel(const u16* __restrict__ k0b, const float* __restrict__ Wu_w,
                  const float* __restrict__ Wu_b, float* __restrict__ uw) {
    const int b = blockIdx.x, h = blockIdx.y, tid = threadIdx.x;
    __shared__ __align__(16) float wu[1024];
    __shared__ float ulog[1024];
    __shared__ float red[256];
    for (int i = tid; i < 1024; i += 256) wu[i] = Wu_w[h*1024 + i];
    __syncthreads();
    const float bias = Wu_b[h];
    for (int li = 0; li < 4; ++li) {
        const int l = tid + li*256;
        const u16* row = k0b + ((size_t)b*1024 + l) * 1024;
        float s = 0.f;
        for (int e = 0; e < 1024; e += 8) {
            const u32x4 pk8 = *(const u32x4*)(row + e);
            #pragma unroll
            for (int j = 0; j < 4; ++j) {
                const u32 p = pk8[j];
                s += bf2f((u16)(p & 0xFFFFu)) * wu[e + j*2];
                s += bf2f((u16)(p >> 16))     * wu[e + j*2 + 1];
            }
        }
        ulog[l] = s + bias;
    }
    __syncthreads();
    float m = fmaxf(fmaxf(ulog[tid], ulog[tid+256]), fmaxf(ulog[tid+512], ulog[tid+768]));
    red[tid] = m; __syncthreads();
    for (int st = 128; st > 0; st >>= 1) {
        if (tid < st) red[tid] = fmaxf(red[tid], red[tid+st]);
        __syncthreads();
    }
    const float mx = red[0];
    __syncthreads();
    float sm = 0.f;
    for (int li = 0; li < 4; ++li) sm += __expf(ulog[tid + li*256] - mx);
    red[tid] = sm; __syncthreads();
    for (int st = 128; st > 0; st >>= 1) {
        if (tid < st) red[tid] += red[tid+st];
        __syncthreads();
    }
    const float inv = 1.f / red[0];
    for (int li = 0; li < 4; ++li) {
        const int l = tid + li*256;
        const float p = __expf(ulog[l] - mx) * inv;
        const int bp = b*8 + (l >> 7);             // faithful row-major reshape
        const int lp = ((l & 127) << 3) + h;
        uw[bp*1024 + lp] = p;
    }
}

// uwv[bh][d] = sum_l uw[bh][l] * v[bh][l][d]
__global__ void uwv_kernel(const float* __restrict__ uw, const float* __restrict__ vf,
                           float* __restrict__ uwv) {
    const int bh = blockIdx.x, d = threadIdx.x;
    const float* u = uw + bh*1024;
    const float* v = vf + (size_t)bh*65536;
    float s = 0.f;
    for (int l = 0; l < 1024; ++l) s += u[l] * v[l*64 + d];
    uwv[bh*64 + d] = s;
}

// ---------------- flash attention: out = softmax(muq@muk^T)@v + uwv ----------------
// (1/sqrt(HD) pre-folded into muq). grid (64 heads, 8 q-blocks of 128), block 256.
__global__ __launch_bounds__(256)
void flash_attn_kernel(const u16* __restrict__ muq, const u16* __restrict__ muk,
                       const u16* __restrict__ vt, const float* __restrict__ uwv,
                       float* __restrict__ out) {
    const int bh = blockIdx.x, qblk = blockIdx.y;
    const int tid = threadIdx.x;
    const int wave = tid >> 5, lane = tid & 31;
    const int lm = lane & 15, half = lane >> 4;
    const int klo8 = half * 8;
    __shared__ __align__(16) u16 ldsP[8][16 * 64];

    const u16* mq = muq + (size_t)bh * 65536;
    const u16* mk = muk + (size_t)bh * 65536;
    const u16* vv = vt  + (size_t)bh * 65536;
    const int qbase = qblk * 128 + wave * 16;

    Frag aq[2];
    {
        const u16* arow = mq + (size_t)(qbase + lm) * 64;
        #pragma unroll
        for (int j = 0; j < 2; ++j) {
            aq[j].lo = *(const u32x4*)(arow + j*32 + klo8);
            aq[j].hi = *(const u32x4*)(arow + j*32 + klo8 + 16);
        }
    }
    f32x8 o[4];
    #pragma unroll
    for (int dn = 0; dn < 4; ++dn) o[dn] = zero8();
    float rmax[8], rsum[8];
    #pragma unroll
    for (int r = 0; r < 8; ++r) { rmax[r] = -3.0e38f; rsum[r] = 0.f; }

    for (int kt = 0; kt < 16; ++kt) {
        const int kb = kt * 64;
        f32x8 s[4];
        #pragma unroll
        for (int n = 0; n < 4; ++n) s[n] = zero8();
        #pragma unroll
        for (int n = 0; n < 4; ++n) {
            const u16* brow = mk + (size_t)(kb + n*16 + lm) * 64 + half*8;
            #pragma unroll
            for (int j = 0; j < 2; ++j) {
                Frag bk;
                bk.lo = *(const u32x4*)(brow + j*32);
                bk.hi = *(const u32x4*)(brow + j*32 + 16);
                s[n] = wmma_bf16(aq[j], bk, s[n]);
            }
        }
        float mnew[8], alpha[8], tsum[8];
        #pragma unroll
        for (int r = 0; r < 8; ++r) {
            float t = fmaxf(fmaxf(s[0][r], s[1][r]), fmaxf(s[2][r], s[3][r]));
            t = hmax16(t);
            mnew[r]  = fmaxf(rmax[r], t);
            alpha[r] = __expf(rmax[r] - mnew[r]);
            rmax[r]  = mnew[r];
            tsum[r]  = 0.f;
        }
        u16* pr = ldsP[wave];
        #pragma unroll
        for (int n = 0; n < 4; ++n)
            #pragma unroll
            for (int r = 0; r < 8; ++r) {
                const float p = __expf(s[n][r] - mnew[r]);
                tsum[r] += p;
                pr[(r + half*8) * 64 + n*16 + lm] = f2bf(p);  // C-layout -> tile
            }
        #pragma unroll
        for (int r = 0; r < 8; ++r) {
            rsum[r] = rsum[r] * alpha[r] + hsum16(tsum[r]);
        }
        #pragma unroll
        for (int dn = 0; dn < 4; ++dn)
            #pragma unroll
            for (int r = 0; r < 8; ++r) o[dn][r] *= alpha[r];
        __syncthreads();
        Frag pa[2];
        {
            const u16* prow = &ldsP[wave][lm * 64];           // re-read as A-fragment
            #pragma unroll
            for (int j = 0; j < 2; ++j) {
                pa[j].lo = *(const u32x4*)(prow + j*32 + klo8);
                pa[j].hi = *(const u32x4*)(prow + j*32 + klo8 + 16);
            }
        }
        #pragma unroll
        for (int dn = 0; dn < 4; ++dn) {
            const u16* vrow = vv + (size_t)(dn*16 + lm) * 1024 + kb + half*8;
            #pragma unroll
            for (int j = 0; j < 2; ++j) {
                Frag bvf;
                bvf.lo = *(const u32x4*)(vrow + j*32);
                bvf.hi = *(const u32x4*)(vrow + j*32 + 16);
                o[dn] = wmma_bf16(pa[j], bvf, o[dn]);
            }
        }
        __syncthreads();
    }
    #pragma unroll
    for (int dn = 0; dn < 4; ++dn) {
        const int d = dn*16 + lm;
        const float uadd = uwv[bh*64 + d];
        #pragma unroll
        for (int r = 0; r < 8; ++r) {
            const int row = qbase + r + half*8;
            out[(size_t)bh*65536 + (size_t)row*64 + d] = o[dn][r] / rsum[r] + uadd;
        }
    }
}

extern "C" void kernel_launch(void* const* d_in, const int* in_sizes, int n_in,
                              void* d_out, int out_size, void* d_ws, size_t ws_size,
                              hipStream_t stream) {
    const float* x    = (const float*)d_in[0];
    const float* W_Q  = (const float*)d_in[1];
    const float* W_K  = (const float*)d_in[2];
    const float* W_V  = (const float*)d_in[3];
    const float* Wq_w = (const float*)d_in[4];
    const float* Wq_b = (const float*)d_in[5];
    const float* Wk_w = (const float*)d_in[6];
    const float* Wk_b = (const float*)d_in[7];
    const float* Wv_w = (const float*)d_in[8];
    const float* Wv_b = (const float*)d_in[9];
    const float* Wu_w = (const float*)d_in[10];
    const float* Wu_b = (const float*)d_in[11];

    size_t off = 0;
    char* base = (char*)d_ws;
    auto alloc = [&](size_t bytes) {
        void* p = base + off;
        off += (bytes + 255) & ~(size_t)255;
        return p;
    };
    u16* xb   = (u16*)alloc((size_t)8192*1024*2);   // x in bf16
    u16* WQt  = (u16*)alloc((size_t)1024*1024*2);   // W_Q^T bf16 [N][K]
    u16* WKt  = (u16*)alloc((size_t)1024*1024*2);
    u16* WVt  = (u16*)alloc((size_t)1024*1024*2);
    u16* Wqt  = (u16*)alloc((size_t)512*1024*2);    // Wq_w bf16 (already [N][K])
    u16* Wkt  = (u16*)alloc((size_t)512*1024*2);
    u16* Wvt  = (u16*)alloc((size_t)512*1024*2);
    u16* q0b  = (u16*)alloc((size_t)8192*1024*2);
    u16* k0b  = (u16*)alloc((size_t)8192*1024*2);
    u16* v0b  = (u16*)alloc((size_t)8192*1024*2);
    float* qf = (float*)alloc((size_t)8192*512*4);
    float* kf = (float*)alloc((size_t)8192*512*4);
    float* vf = (float*)alloc((size_t)8192*512*4);
    u16* muq  = (u16*)alloc((size_t)64*1024*64*2);
    u16* muk  = (u16*)alloc((size_t)64*1024*64*2);
    u16* vtb  = (u16*)alloc((size_t)64*64*1024*2);
    float* uw  = (float*)alloc((size_t)64*1024*4);
    float* uwv = (float*)alloc((size_t)64*64*4);

    // 1) converts
    cvt_bf16_kernel<<<8388608/256, 256, 0, stream>>>(xb, x, 8388608);
    tcvt_bf16_kernel<<<1048576/256, 256, 0, stream>>>(WQt, W_Q, 1024, 1024);
    tcvt_bf16_kernel<<<1048576/256, 256, 0, stream>>>(WKt, W_K, 1024, 1024);
    tcvt_bf16_kernel<<<1048576/256, 256, 0, stream>>>(WVt, W_V, 1024, 1024);
    cvt_bf16_kernel<<<524288/256, 256, 0, stream>>>(Wqt, Wq_w, 524288);
    cvt_bf16_kernel<<<524288/256, 256, 0, stream>>>(Wkt, Wk_w, 524288);
    cvt_bf16_kernel<<<524288/256, 256, 0, stream>>>(Wvt, Wv_w, 524288);

    // 2) full-width projections (bf16 out)
    gemm_bf16_kernel<true,false><<<dim3(64,8), 256, 0, stream>>>(xb, WQt, nullptr, q0b, 8192, 1024, 1024);
    gemm_bf16_kernel<true,false><<<dim3(64,8), 256, 0, stream>>>(xb, WKt, nullptr, k0b, 8192, 1024, 1024);
    gemm_bf16_kernel<true,false><<<dim3(64,8), 256, 0, stream>>>(xb, WVt, nullptr, v0b, 8192, 1024, 1024);

    // 3) attention-dim projections + bias + relu (f32 out)
    gemm_bf16_kernel<false,true><<<dim3(64,4), 256, 0, stream>>>(q0b, Wqt, Wq_b, qf, 8192, 512, 1024);
    gemm_bf16_kernel<false,true><<<dim3(64,4), 256, 0, stream>>>(k0b, Wkt, Wk_b, kf, 8192, 512, 1024);
    gemm_bf16_kernel<false,true><<<dim3(64,4), 256, 0, stream>>>(v0b, Wvt, Wv_b, vf, 8192, 512, 1024);

    // 4) mean centering (per-head, folds 1/sqrt(HD) into muq) + v transpose
    center_kernel<<<64, 256, 0, stream>>>(qf, kf, vf, muq, muk, vtb);

    // 5) unary softmax bias -> uw[64][1024], then uwv[64][64]
    unary_kernel<<<dim3(8,8), 256, 0, stream>>>(k0b, Wu_w, Wu_b, uw);
    uwv_kernel<<<64, 64, 0, stream>>>(uw, vf, uwv);

    // 6) flash attention, writes final [B,L,A] (flat identity with [64][1024][64])
    flash_attn_kernel<<<dim3(64,8), 256, 0, stream>>>(muq, muk, vtb, uwv, (float*)d_out);
}
